// TransformerBlock_6107443495169
// MI455X (gfx1250) — matrix operand order repaired and checked by reading, not compiled, not run on hardware
//
#include <hip/hip_runtime.h>

// ---------------------------------------------------------------------------
// Types for CDNA5 WMMA (wave32): v_wmma_f32_16x16x32_bf16
// ---------------------------------------------------------------------------
typedef __attribute__((ext_vector_type(16))) __bf16 v16bf;
typedef __attribute__((ext_vector_type(8)))  __bf16 v8bf;
typedef __attribute__((ext_vector_type(8)))  float  v8f;

__device__ __forceinline__ __bf16 cvt_bf16(float v)  { return (__bf16)v; }
__device__ __forceinline__ __bf16 cvt_bf16(__bf16 v) { return v; }

__device__ __forceinline__ v16bf load_frag(const __bf16* lo, const __bf16* hi)
{
    v8bf a = *(const v8bf*)lo;
    v8bf b = *(const v8bf*)hi;
    return __builtin_shufflevector(a, b, 0,1,2,3,4,5,6,7,8,9,10,11,12,13,14,15);
}

// ---------------------------------------------------------------------------
// Tiled WMMA GEMM:  C[M,N] = act( scale * (A @ B) + bias )
//   A: row-major [M,K], type AT (float or __bf16), leading dim lda
//   B: if !BTRANS: B[k][n] = Bg[k*ldb + n]   (NN)
//      if  BTRANS: B[k][n] = Bg[n*ldb + k]   (NT, e.g. Q @ K^T)
//   C: row-major [M,N], type OT (float or __bf16), leading dim ldc
// Block: 256 threads = 8 waves; 128x64 C tile; per-wave 32x32; K stepped by 32.
// Software-pipelined with ping-pong LDS:
//   bf16 operands prefetched with GLOBAL_LOAD_ASYNC_TO_LDS_B128 (ASYNCcnt),
//   fp32 operands loaded into regs early, cvt+ds_store committed after WMMA.
// causal_skip:   skip blocks fully above the diagonal (softmax re-masks)
// causal_klimit: clamp K loop to m0+128 (masked probs are exactly 0)
// Requires M%128==0, N%64==0, K%32==0 (true for all shapes here).
// ---------------------------------------------------------------------------
template <typename AT, typename BT, bool BTRANS, bool HAS_BIAS, bool RELU, typename OT>
__global__ __launch_bounds__(256)
void gemm_wmma(const AT* __restrict__ A, const BT* __restrict__ Bg,
               const float* __restrict__ bias, OT* __restrict__ C,
               int M, int N, int K, int lda, int ldb, int ldc,
               float scale, int causal_skip, int causal_klimit)
{
    constexpr int BM = 128, BN = 64, LDS_S = 40;  // 80B LDS rows (16B aligned)
    constexpr bool ASYNC_A = (sizeof(AT) == 2);
    constexpr bool ASYNC_B = BTRANS && (sizeof(BT) == 2);
    __shared__ __align__(16) __bf16 As[2][BM * LDS_S];
    __shared__ __align__(16) __bf16 Bs[2][BN * LDS_S];

    const int t    = threadIdx.x;
    const int lane = t & 31;
    const int wid  = t >> 5;       // 0..7
    const int wn   = wid & 1;      // 2 N subtiles of 16 each
    const int wm   = wid >> 1;     // 0..3, 2 M subtiles of 16 each
    const int half = lane >> 4;
    const int ln   = lane & 15;

    const int m0 = blockIdx.y * BM;
    const int n0 = blockIdx.x * BN;

    if (causal_skip && n0 > m0 + (BM - 1)) return;   // fully-masked score block

    v8f acc00 = {}, acc01 = {}, acc10 = {}, acc11 = {};

    int Keff = K;
    if (causal_klimit && (m0 + BM) < K) Keff = m0 + BM;  // keys beyond row have P==0
    const int ksteps = Keff >> 5;

    AT ra[16];            // fp32-A staging regs (4 passes x 4)
    BT rb[8];             // B staging regs (2 passes x 4)

    // ---- stage issue: global loads (async -> LDS, or into regs) ----
    auto issueA = [&](int k0, int buf) {
        if constexpr (ASYNC_A) {
            #pragma unroll
            for (int p = 0; p < 2; ++p) {
                int idx = t + p * 256;           // 512 chunks of 16B
                int row = idx >> 2, seg = idx & 3;
                const void* g = (const void*)(A + (size_t)(m0 + row) * lda + k0 + seg * 8);
                unsigned l = (unsigned)(size_t)&As[buf][row * LDS_S + seg * 8];
                asm volatile("global_load_async_to_lds_b128 %0, %1, off"
                             :: "v"(l), "v"(g) : "memory");
            }
        } else {
            #pragma unroll
            for (int p = 0; p < 4; ++p) {
                int idx = t + p * 256;           // 1024 groups of 4
                int row = idx >> 3, c4 = (idx & 7) << 2;
                const AT* src = A + (size_t)(m0 + row) * lda + k0 + c4;
                #pragma unroll
                for (int j = 0; j < 4; ++j) ra[p * 4 + j] = src[j];
            }
        }
    };
    auto commitA = [&](int buf) {      // fp32 path only: cvt + ds_store
        if constexpr (!ASYNC_A) {
            #pragma unroll
            for (int p = 0; p < 4; ++p) {
                int idx = t + p * 256;
                int row = idx >> 3, c4 = (idx & 7) << 2;
                __bf16* dst = &As[buf][row * LDS_S + c4];
                #pragma unroll
                for (int j = 0; j < 4; ++j) dst[j] = cvt_bf16(ra[p * 4 + j]);
            }
        }
    };
    auto issueB = [&](int k0, int buf) {
        if constexpr (ASYNC_B) {
            int row = t >> 2, seg = t & 3;       // 256 chunks of 16B
            const void* g = (const void*)(Bg + (size_t)(n0 + row) * ldb + k0 + seg * 8);
            unsigned l = (unsigned)(size_t)&Bs[buf][row * LDS_S + seg * 8];
            asm volatile("global_load_async_to_lds_b128 %0, %1, off"
                         :: "v"(l), "v"(g) : "memory");
        } else if constexpr (BTRANS) {
            #pragma unroll
            for (int p = 0; p < 2; ++p) {
                int idx = t + p * 256;           // 512 groups of 4 (contiguous k)
                int row = idx >> 3, c4 = (idx & 7) << 2;
                const BT* src = Bg + (size_t)(n0 + row) * ldb + k0 + c4;
                #pragma unroll
                for (int j = 0; j < 4; ++j) rb[p * 4 + j] = src[j];
            }
        } else {
            #pragma unroll
            for (int p = 0; p < 2; ++p) {
                int idx = t + p * 256;           // 512: 32 kk x 16 n-groups
                int kk = idx >> 4, n4 = (idx & 15) << 2;
                const BT* src = Bg + (size_t)(k0 + kk) * ldb + n0 + n4;
                #pragma unroll
                for (int j = 0; j < 4; ++j) rb[p * 4 + j] = src[j];
            }
        }
    };
    auto commitB = [&](int buf) {
        if constexpr (ASYNC_B) {
        } else if constexpr (BTRANS) {
            #pragma unroll
            for (int p = 0; p < 2; ++p) {
                int idx = t + p * 256;
                int row = idx >> 3, c4 = (idx & 7) << 2;
                __bf16* dst = &Bs[buf][row * LDS_S + c4];
                #pragma unroll
                for (int j = 0; j < 4; ++j) dst[j] = cvt_bf16(rb[p * 4 + j]);
            }
        } else {
            #pragma unroll
            for (int p = 0; p < 2; ++p) {
                int idx = t + p * 256;
                int kk = idx >> 4, n4 = (idx & 15) << 2;
                #pragma unroll
                for (int j = 0; j < 4; ++j)
                    Bs[buf][(n4 + j) * LDS_S + kk] = cvt_bf16(rb[p * 4 + j]);
            }
        }
    };

    // ---- prologue: stage tile 0 into buffer 0 ----
    issueA(0, 0);
    issueB(0, 0);
    commitA(0);
    commitB(0);
    if constexpr (ASYNC_A || ASYNC_B)
        asm volatile("s_wait_asynccnt 0" ::: "memory");
    __syncthreads();

    int cur = 0;
    for (int kt = 0; kt < ksteps; ++kt) {
        const int nxt = cur ^ 1;
        const bool have_next = (kt + 1) < ksteps;
        // prefetch next tile: async copies go straight to LDS[nxt];
        // fp32 loads are issued now (latency hidden behind the WMMAs below)
        if (have_next) {
            issueA((kt + 1) << 5, nxt);
            issueB((kt + 1) << 5, nxt);
        }

        // ---- fragments from LDS[cur] (contiguous-k order, same for A and B) ----
        // B: N = lane%16, K = half*16 + i (contiguous 16)
        const __bf16* Bbase = Bs[cur];
        const __bf16* Abase = As[cur];
        const __bf16* bp0 = &Bbase[((wn * 2 + 0) * 16 + ln) * LDS_S + half * 16];
        const __bf16* bp1 = &Bbase[((wn * 2 + 1) * 16 + ln) * LDS_S + half * 16];
        v16bf bf0 = load_frag(bp0, bp0 + 8);
        v16bf bf1 = load_frag(bp1, bp1 + 8);
        // A: M = lane%16, K = half*16 + i (contiguous 16)
        const __bf16* ap0 = &Abase[((wm * 2 + 0) * 16 + ln) * LDS_S + half * 16];
        const __bf16* ap1 = &Abase[((wm * 2 + 1) * 16 + ln) * LDS_S + half * 16];
        v16bf af0 = load_frag(ap0, ap0 + 8);
        v16bf af1 = load_frag(ap1, ap1 + 8);

        acc00 = __builtin_amdgcn_wmma_f32_16x16x32_bf16(false, af0, false, bf0, (short)0, acc00, false, false);
        acc01 = __builtin_amdgcn_wmma_f32_16x16x32_bf16(false, af0, false, bf1, (short)0, acc01, false, false);
        acc10 = __builtin_amdgcn_wmma_f32_16x16x32_bf16(false, af1, false, bf0, (short)0, acc10, false, false);
        acc11 = __builtin_amdgcn_wmma_f32_16x16x32_bf16(false, af1, false, bf1, (short)0, acc11, false, false);

        // commit reg-staged data to LDS[nxt] (wait for global loads lands here,
        // i.e. after the matrix work, not before it)
        if (have_next) {
            commitA(nxt);
            commitB(nxt);
        }
        if constexpr (ASYNC_A || ASYNC_B)
            asm volatile("s_wait_asynccnt 0" ::: "memory");
        __syncthreads();
        cur = nxt;
    }

    // ---- epilogue: D layout VGPR r -> M = r + 8*half, N = lane%16 ----
    const int col0 = n0 + (wn * 2 + 0) * 16 + ln;
    const int col1 = n0 + (wn * 2 + 1) * 16 + ln;
    float bv0 = 0.f, bv1 = 0.f;
    if constexpr (HAS_BIAS) { bv0 = bias[col0]; bv1 = bias[col1]; }

    auto store_tile = [&](const v8f& acc, int rbase, int col, float bv) {
        #pragma unroll
        for (int r = 0; r < 8; ++r) {
            float v = acc[r] * scale;
            if constexpr (HAS_BIAS) v += bv;
            if constexpr (RELU) v = v > 0.f ? v : 0.f;
            size_t o = (size_t)(rbase + r) * ldc + col;
            if constexpr (sizeof(OT) == 2) C[o] = (__bf16)v;
            else                           C[o] = v;
        }
    };
    const int rb0 = m0 + (wm * 2 + 0) * 16 + half * 8;
    const int rb1 = m0 + (wm * 2 + 1) * 16 + half * 8;
    store_tile(acc00, rb0, col0, bv0);
    store_tile(acc01, rb0, col1, bv1);
    store_tile(acc10, rb1, col0, bv0);
    store_tile(acc11, rb1, col1, bv1);
}

// ---------------------------------------------------------------------------
// Row softmax with optional causal mask; fp32 scores in, bf16 probs out.
// One block (256 threads) per row; L = 2048 -> 8 cols/thread.
// ---------------------------------------------------------------------------
__global__ __launch_bounds__(256)
void softmax_rows(const float* __restrict__ S, __bf16* __restrict__ P,
                  int L, int causal)
{
    __shared__ float red[256];
    const int row = blockIdx.x;
    const int t = threadIdx.x;
    const int C = L >> 8;           // 8
    float v[8];
    const float* srow = S + (size_t)row * L;
    float mx = -3.0e38f;
    for (int i = 0; i < C; ++i) {
        int c = t + (i << 8);
        float x = srow[c];
        if (causal && c > row) x = -3.0e38f;
        v[i] = x;
        mx = fmaxf(mx, x);
    }
    red[t] = mx; __syncthreads();
    for (int s = 128; s > 0; s >>= 1) {
        if (t < s) red[t] = fmaxf(red[t], red[t + s]);
        __syncthreads();
    }
    mx = red[0]; __syncthreads();
    float sum = 0.f;
    for (int i = 0; i < C; ++i) { v[i] = __expf(v[i] - mx); sum += v[i]; }
    red[t] = sum; __syncthreads();
    for (int s = 128; s > 0; s >>= 1) {
        if (t < s) red[t] += red[t + s];
        __syncthreads();
    }
    float inv = 1.0f / red[0];
    __bf16* prow = P + (size_t)row * L;
    for (int i = 0; i < C; ++i) prow[t + (i << 8)] = (__bf16)(v[i] * inv);
}

// ---------------------------------------------------------------------------
// out = LayerNorm(a + b) * gamma + beta  (eps = 1e-3, fp32)
// One block (256 threads) per row; D = 1024 -> 4 cols/thread.
// ---------------------------------------------------------------------------
__global__ __launch_bounds__(256)
void add_layernorm(const float* __restrict__ a, const float* __restrict__ b,
                   const float* __restrict__ gamma, const float* __restrict__ beta,
                   float* __restrict__ out, int D)
{
    __shared__ float red[256];
    const int row = blockIdx.x;
    const int t = threadIdx.x;
    const int C = D >> 8;           // 4
    float v[4];
    const float* ar = a + (size_t)row * D;
    const float* br = b + (size_t)row * D;
    float s = 0.f;
    for (int i = 0; i < C; ++i) {
        int c = t + (i << 8);
        v[i] = ar[c] + br[c];
        s += v[i];
    }
    red[t] = s; __syncthreads();
    for (int k = 128; k > 0; k >>= 1) {
        if (t < k) red[t] += red[t + k];
        __syncthreads();
    }
    float mu = red[0] / D; __syncthreads();
    float sq = 0.f;
    for (int i = 0; i < C; ++i) { float d = v[i] - mu; sq += d * d; }
    red[t] = sq; __syncthreads();
    for (int k = 128; k > 0; k >>= 1) {
        if (t < k) red[t] += red[t + k];
        __syncthreads();
    }
    float inv = rsqrtf(red[0] / D + 1e-3f);
    float* orow = out + (size_t)row * D;
    for (int i = 0; i < C; ++i) {
        int c = t + (i << 8);
        orow[c] = (v[i] - mu) * inv * gamma[c] + beta[c];
    }
}

// ---------------------------------------------------------------------------
// Host side: full transformer block pipeline
// ---------------------------------------------------------------------------
extern "C" void kernel_launch(void* const* d_in, const int* in_sizes, int n_in,
                              void* d_out, int out_size, void* d_ws, size_t ws_size,
                              hipStream_t stream)
{
    (void)in_sizes; (void)n_in; (void)out_size; (void)ws_size;

    const float* X     = (const float*)d_in[0];   // [8,2048,1024]
    const float* CTX   = (const float*)d_in[1];   // [8,2048,1024]
    const float* WkS   = (const float*)d_in[2];
    const float* WvS   = (const float*)d_in[3];
    const float* WqS   = (const float*)d_in[4];
    const float* WkC   = (const float*)d_in[5];
    const float* WvC   = (const float*)d_in[6];
    const float* WqC   = (const float*)d_in[7];
    const float* W1    = (const float*)d_in[8];   // [1024,2048]
    const float* b1    = (const float*)d_in[9];
    const float* W2    = (const float*)d_in[10];  // [2048,1024]
    const float* b2    = (const float*)d_in[11];
    const float* gamma = (const float*)d_in[12];
    const float* beta  = (const float*)d_in[13];
    float* OUT = (float*)d_out;

    constexpr int NB = 8, L = 2048, D = 1024, H = 2048;
    constexpr int M = NB * L;                       // 16384 rows
    const float scl = 0.022097086912079608f;        // 1/sqrt(2048) == 1/sqrt(Lk)

    // workspace layout (~327 MB)
    char* ws = (char*)d_ws;
    size_t off = 0;
    auto alloc = [&](size_t bytes) -> char* {
        char* p = ws + off;
        off += (bytes + 255) & ~(size_t)255;
        return p;
    };
    __bf16* bufQ = (__bf16*)alloc((size_t)M * D * 2);   // 32 MB
    __bf16* bufK = (__bf16*)alloc((size_t)M * D * 2);   // 32 MB
    __bf16* bufV = (__bf16*)alloc((size_t)M * D * 2);   // 32 MB
    float*  S    = (float *)alloc((size_t)L * L * 4);   // 16 MB (per-batch scores)
    __bf16* P    = (__bf16*)alloc((size_t)L * L * 2);   //  8 MB (per-batch probs)
    float*  AO   = (float *)alloc((size_t)M * D * 4);   // 64 MB (attn / ffn out)
    float*  X1   = (float *)alloc((size_t)M * D * 4);   // 64 MB
    float*  X2   = (float *)alloc((size_t)M * D * 4);   // 64 MB
    __bf16* Hbuf = (__bf16*)bufQ;  // FFN hidden [M,H] bf16 = 64 MB, aliases Q+K exactly

    dim3 blk(256);
    dim3 gProj(D / 64, M / 128);    // [16384,1024] x [1024,1024]
    dim3 gScore(L / 64, L / 128);   // per batch: [2048,2048]
    dim3 gPV(D / 64, L / 128);      // per batch: [2048,1024]
    dim3 gFF1(H / 64, M / 128);
    dim3 gFF2(D / 64, M / 128);

    // ---- self-attention projections (fp32 in -> bf16 out) ----
    gemm_wmma<float, float, false, false, false, __bf16>
        <<<gProj, blk, 0, stream>>>(X, WkS, nullptr, bufK, M, D, D, D, D, D, 1.0f, 0, 0);
    gemm_wmma<float, float, false, false, false, __bf16>
        <<<gProj, blk, 0, stream>>>(X, WvS, nullptr, bufV, M, D, D, D, D, D, 1.0f, 0, 0);
    gemm_wmma<float, float, false, false, false, __bf16>
        <<<gProj, blk, 0, stream>>>(X, WqS, nullptr, bufQ, M, D, D, D, D, D, 1.0f, 0, 0);

    // ---- causal self-attention, per batch ----
    for (int b = 0; b < NB; ++b) {
        const __bf16* Qb = bufQ + (size_t)b * L * D;
        const __bf16* Kb = bufK + (size_t)b * L * D;
        const __bf16* Vb = bufV + (size_t)b * L * D;
        float* Ob = AO + (size_t)b * L * D;
        gemm_wmma<__bf16, __bf16, true, false, false, float>
            <<<gScore, blk, 0, stream>>>(Qb, Kb, nullptr, S, L, L, D, D, D, L, scl, 1, 0);
        softmax_rows<<<dim3(L), blk, 0, stream>>>(S, P, L, 1);
        gemm_wmma<__bf16, __bf16, false, false, false, float>
            <<<gPV, blk, 0, stream>>>(P, Vb, nullptr, Ob, L, D, L, L, D, D, 1.0f, 0, 1);
    }
    add_layernorm<<<dim3(M), blk, 0, stream>>>(X, AO, gamma, beta, X1, D);

    // ---- cross-attention projections ----
    gemm_wmma<float, float, false, false, false, __bf16>
        <<<gProj, blk, 0, stream>>>(CTX, WkC, nullptr, bufK, M, D, D, D, D, D, 1.0f, 0, 0);
    gemm_wmma<float, float, false, false, false, __bf16>
        <<<gProj, blk, 0, stream>>>(CTX, WvC, nullptr, bufV, M, D, D, D, D, D, 1.0f, 0, 0);
    gemm_wmma<float, float, false, false, false, __bf16>
        <<<gProj, blk, 0, stream>>>(X1, WqC, nullptr, bufQ, M, D, D, D, D, D, 1.0f, 0, 0);

    // ---- cross-attention (no mask), per batch ----
    for (int b = 0; b < NB; ++b) {
        const __bf16* Qb = bufQ + (size_t)b * L * D;
        const __bf16* Kb = bufK + (size_t)b * L * D;
        const __bf16* Vb = bufV + (size_t)b * L * D;
        float* Ob = AO + (size_t)b * L * D;
        gemm_wmma<__bf16, __bf16, true, false, false, float>
            <<<gScore, blk, 0, stream>>>(Qb, Kb, nullptr, S, L, L, D, D, D, L, scl, 0, 0);
        softmax_rows<<<dim3(L), blk, 0, stream>>>(S, P, L, 0);
        gemm_wmma<__bf16, __bf16, false, false, false, float>
            <<<gPV, blk, 0, stream>>>(P, Vb, nullptr, Ob, L, D, L, L, D, D, 1.0f, 0, 0);
    }
    add_layernorm<<<dim3(M), blk, 0, stream>>>(X1, AO, gamma, beta, X2, D);

    // ---- FFN: relu(X2 @ W1 + b1) @ W2 + b2 ----
    gemm_wmma<float, float, false, true, true, __bf16>
        <<<gFF1, blk, 0, stream>>>(X2, W1, b1, Hbuf, M, H, D, D, H, H, 1.0f, 0, 0);
    gemm_wmma<__bf16, float, false, true, false, float>
        <<<gFF2, blk, 0, stream>>>(Hbuf, W2, b2, AO, M, D, H, H, D, D, 1.0f, 0, 0);

    add_layernorm<<<dim3(M), blk, 0, stream>>>(X2, AO, gamma, beta, OUT, D);
}
